// DynamicAttention21_40827959116535
// MI455X (gfx1250) — compile-verified, matrix operand order checked by reading, not run on hardware
//
#include <hip/hip_runtime.h>

// ---------------------------------------------------------------------------
// Types / WMMA helpers (CDNA5 wave32, v_wmma_f32_16x16x32_f16)
// ---------------------------------------------------------------------------
typedef __attribute__((ext_vector_type(16))) _Float16 v16h;
typedef __attribute__((ext_vector_type(8)))  _Float16 v8h;
typedef __attribute__((ext_vector_type(8)))  float    v8f;

__device__ __forceinline__ v8f wmma16(v16h a, v16h b, v8f c) {
  // (neg_a, A, neg_b, B, c_mod, C, reuse_a, reuse_b)
  return __builtin_amdgcn_wmma_f32_16x16x32_f16(false, a, false, b, (short)0, c,
                                                false, false);
}

// A/B fragment (16x32 f16, M/N x K) from a K-contiguous matrix [row][K].
// Lane L<16 : row = row0+L,     K = {k0..k0+7, k0+16..k0+23}
// Lane L>=16: row = row0+L-16,  K shifted by +8
// => two contiguous 128-bit loads per lane.
__device__ __forceinline__ v16h load_frag(const _Float16* base, int ld,
                                          int row0, int k0, int lane) {
  const int r  = row0 + (lane & 15);
  const int kk = k0 + ((lane >> 4) << 3);
  const _Float16* p = base + (size_t)r * ld + kk;
  v8h lo = *(const v8h*)p;
  v8h hi = *(const v8h*)(p + 16);
  v16h o;
#pragma unroll
  for (int i = 0; i < 8; ++i) { o[i] = lo[i]; o[i + 8] = hi[i]; }
  return o;
}

// Broadcast fragment: every "row" of A gets the same vector q[k] (for P = q⊙s_key).
__device__ __forceinline__ v16h load_bcast(const _Float16* row, int k0, int lane) {
  const int kk = k0 + ((lane >> 4) << 3);
  v8h lo = *(const v8h*)(row + kk);
  v8h hi = *(const v8h*)(row + kk + 16);
  v16h o;
#pragma unroll
  for (int i = 0; i < 8; ++i) { o[i] = lo[i]; o[i + 8] = hi[i]; }
  return o;
}

// ---------------------------------------------------------------------------
// Elementwise conversion / transpose kernels
// ---------------------------------------------------------------------------
__global__ void k_cvt_f16(const float* __restrict__ in, _Float16* __restrict__ out,
                          int rows, int cols, int ldo) {
  int idx = blockIdx.x * blockDim.x + threadIdx.x;
  if (idx >= rows * cols) return;
  int r = idx / cols, c = idx - r * cols;
  out[(size_t)r * ldo + c] = (_Float16)in[idx];
}

// in[K][N] (f32, fan_in-major) -> out[N][K] (f16)
__global__ void k_transpose_w(const float* __restrict__ in, _Float16* __restrict__ out,
                              int K, int N) {
  int idx = blockIdx.x * blockDim.x + threadIdx.x;
  if (idx >= K * N) return;
  int n = idx / K, k = idx - n * K;
  out[idx] = (_Float16)in[(size_t)k * N + n];
}

// tv16[(b*64+t)][512] -> tvT[b][k][t]   (so G-GEMM B operand is [N=k][K=t])
__global__ void k_transpose_tv(const _Float16* __restrict__ tv,
                               _Float16* __restrict__ tvT) {
  int idx = blockIdx.x * blockDim.x + threadIdx.x;  // 16*512*64
  if (idx >= 16 * 512 * 64) return;
  int b = idx >> 15, rem = idx & 32767;
  int k = rem >> 6, t = rem & 63;
  tvT[idx] = tv[((size_t)(b * 64 + t)) * 512 + k];
}

// ---------------------------------------------------------------------------
// Generic WMMA GEMM:  C[M][N] = A[M][K](f16) * BT[N][K](f16)^T + bias
// Block: 128 threads (4 waves). Block tile 64(M) x 128(N); wave tile 32x64.
// C32!=null -> f32 store (+optional relu), else f16 store.
// ---------------------------------------------------------------------------
__global__ __launch_bounds__(128) void k_gemm(const _Float16* __restrict__ A, int lda,
                                              const _Float16* __restrict__ BT, int ldb,
                                              const float* __restrict__ bias,
                                              _Float16* __restrict__ C16,
                                              float* __restrict__ C32,
                                              int ldc, int K, int relu) {
  const int lane = threadIdx.x & 31;
  const int w    = threadIdx.x >> 5;
  const int m_off = blockIdx.y * 64  + 32 * (w & 1);
  const int n_off = blockIdx.x * 128 + 64 * (w >> 1);

  v8f acc[2][4];
#pragma unroll
  for (int i = 0; i < 2; ++i)
#pragma unroll
    for (int j = 0; j < 4; ++j) acc[i][j] = (v8f)0.f;

  for (int k0 = 0; k0 < K; k0 += 32) {
    v16h a0 = load_frag(A, lda, m_off,      k0, lane);
    v16h a1 = load_frag(A, lda, m_off + 16, k0, lane);
#pragma unroll
    for (int j = 0; j < 4; ++j) {
      v16h b = load_frag(BT, ldb, n_off + 16 * j, k0, lane);
      acc[0][j] = wmma16(a0, b, acc[0][j]);
      acc[1][j] = wmma16(a1, b, acc[1][j]);
    }
  }

  const int half = lane >> 4;
#pragma unroll
  for (int i = 0; i < 2; ++i)
#pragma unroll
    for (int j = 0; j < 4; ++j) {
      const int n = n_off + 16 * j + (lane & 15);
      const float bv = bias[n];
#pragma unroll
      for (int v = 0; v < 8; ++v) {
        const int m = m_off + 16 * i + v + 8 * half;
        float val = acc[i][j][v] + bv;
        if (C32) C32[(size_t)m * ldc + n] = relu ? fmaxf(val, 0.f) : val;
        else     C16[(size_t)m * ldc + n] = (_Float16)val;
      }
    }
}

// ---------------------------------------------------------------------------
// Fused trilinear attention per (b,l):
//   W[s,t] = (1/sqrt(512)) * sum_k q[k]*skey[s,k]*tkey[t,k]   (WMMA, K=512)
//   softmax over all 4096 (block-wide)
//   G[s,k] = sum_t W[s,t]*tv[t,k]                              (WMMA, K=64)
//   ctx[k] = sum_s sv[s,k]*G[s,k]  -> x16[.., 1024+k] (f16)
// 4 waves: phase1 each wave owns an s-strip of 16; phase2 each wave owns a
// 128-wide k-chunk. W kept in LDS between phases.
// ---------------------------------------------------------------------------
__global__ __launch_bounds__(128) void k_attn(const _Float16* __restrict__ q16,
                                              const _Float16* __restrict__ skey,
                                              const _Float16* __restrict__ tkey,
                                              const _Float16* __restrict__ sv,
                                              const _Float16* __restrict__ tvT,
                                              _Float16* __restrict__ x16) {
  const int l = blockIdx.x, b = blockIdx.y;
  const int lane = threadIdx.x & 31;
  const int w    = threadIdx.x >> 5;
  const int half = lane >> 4;

  __shared__ _Float16 Wsm[64 * 64];
  __shared__ float    red[8];

  const _Float16* qrow   = q16  + ((size_t)(b * 128 + l)) * 512;
  const _Float16* skey_b = skey + ((size_t)b * 64) * 512;
  const _Float16* tkey_b = tkey + ((size_t)b * 64) * 512;
  const _Float16* sv_b   = sv   + ((size_t)b * 64) * 512;
  const _Float16* tvT_b  = tvT  + ((size_t)b * 512) * 64;

  // ---- phase 1: logits, wave w owns s in [16w, 16w+16), all 64 t ----
  const int s0 = 16 * w;
  v8f wacc[4];
#pragma unroll
  for (int j = 0; j < 4; ++j) wacc[j] = (v8f)0.f;

  for (int k0 = 0; k0 < 512; k0 += 32) {
    v16h a  = load_frag(skey_b, 512, s0, k0, lane);
    v16h qf = load_bcast(qrow, k0, lane);
    a = a * qf;  // P[s,k] = q[k]*s_key[s,k]  (v_pk_mul_f16)
#pragma unroll
    for (int j = 0; j < 4; ++j) {
      v16h bf = load_frag(tkey_b, 512, 16 * j, k0, lane);
      wacc[j] = wmma16(a, bf, wacc[j]);
    }
  }

  // ---- joint softmax over all 4096 ----
  const float scale = 0.044194173824159216f;  // 1/sqrt(512)
  float mx = -3.4e38f;
#pragma unroll
  for (int j = 0; j < 4; ++j)
#pragma unroll
    for (int v = 0; v < 8; ++v) {
      wacc[j][v] *= scale;
      mx = fmaxf(mx, wacc[j][v]);
    }
#pragma unroll
  for (int off = 16; off; off >>= 1) mx = fmaxf(mx, __shfl_xor(mx, off, 32));
  if (lane == 0) red[w] = mx;
  __syncthreads();
  const float gmx = fmaxf(fmaxf(red[0], red[1]), fmaxf(red[2], red[3]));

  float sum = 0.f;
#pragma unroll
  for (int j = 0; j < 4; ++j)
#pragma unroll
    for (int v = 0; v < 8; ++v) {
      float e = __expf(wacc[j][v] - gmx);
      wacc[j][v] = e;
      sum += e;
    }
#pragma unroll
  for (int off = 16; off; off >>= 1) sum += __shfl_xor(sum, off, 32);
  if (lane == 0) red[4 + w] = sum;
  __syncthreads();
  const float inv = 1.f / (red[4] + red[5] + red[6] + red[7]);

#pragma unroll
  for (int j = 0; j < 4; ++j)
#pragma unroll
    for (int v = 0; v < 8; ++v) {
      const int s = s0 + v + 8 * half;
      const int t = 16 * j + (lane & 15);
      Wsm[s * 64 + t] = (_Float16)(wacc[j][v] * inv);
    }
  __syncthreads();

  // ---- phase 2: G = W @ tvT^T, wave w owns k in [128w, 128w+128) ----
  const int kb = 128 * w;
  float ctx[8];
#pragma unroll
  for (int nt = 0; nt < 8; ++nt) ctx[nt] = 0.f;

  for (int mt = 0; mt < 4; ++mt) {      // s strips of 16
    v8f g[8];
#pragma unroll
    for (int nt = 0; nt < 8; ++nt) g[nt] = (v8f)0.f;
#pragma unroll
    for (int kt = 0; kt < 64; kt += 32) {  // over t
      v16h a = load_frag(Wsm, 64, 16 * mt, kt, lane);
#pragma unroll
      for (int nt = 0; nt < 8; ++nt) {
        v16h bf = load_frag(tvT_b, 64, kb + 16 * nt, kt, lane);
        g[nt] = wmma16(a, bf, g[nt]);
      }
    }
    // fold s-reduction against s_value
#pragma unroll
    for (int nt = 0; nt < 8; ++nt) {
      const int kg = kb + 16 * nt + (lane & 15);
#pragma unroll
      for (int v = 0; v < 8; ++v) {
        const int s = 16 * mt + v + 8 * half;
        ctx[nt] += g[nt][v] * (float)sv_b[(size_t)s * 512 + kg];
      }
    }
  }
#pragma unroll
  for (int nt = 0; nt < 8; ++nt) ctx[nt] += __shfl_xor(ctx[nt], 16, 32);

  if (lane < 16) {
    _Float16* xrow = x16 + ((size_t)(b * 128 + l)) * 1536;
#pragma unroll
    for (int nt = 0; nt < 8; ++nt)
      xrow[1024 + kb + 16 * nt + lane] = (_Float16)ctx[nt];
  }
}

// ---------------------------------------------------------------------------
// Host launch
// ---------------------------------------------------------------------------
extern "C" void kernel_launch(void* const* d_in, const int* in_sizes, int n_in,
                              void* d_out, int out_size, void* d_ws, size_t ws_size,
                              hipStream_t stream) {
  (void)in_sizes; (void)n_in; (void)out_size; (void)ws_size;
  const float* query = (const float*)d_in[0];
  const float* src   = (const float*)d_in[1];
  const float* trg   = (const float*)d_in[2];
  const float* Wq  = (const float*)d_in[3];  const float* bq  = (const float*)d_in[4];
  const float* Ws  = (const float*)d_in[5];  const float* bs  = (const float*)d_in[6];
  const float* Wt  = (const float*)d_in[7];  const float* bt  = (const float*)d_in[8];
  const float* Wsv = (const float*)d_in[9];  const float* bsv = (const float*)d_in[10];
  const float* Wtv = (const float*)d_in[11]; const float* btv = (const float*)d_in[12];
  const float* Wo  = (const float*)d_in[13]; const float* bo  = (const float*)d_in[14];
  float* out = (float*)d_out;

  char* ws = (char*)d_ws;
  size_t off = 0;
  auto alloc = [&](size_t bytes) { char* p = ws + off; off += (bytes + 255) & ~size_t(255); return p; };
  _Float16* x16   = (_Float16*)alloc(2048ull * 1536 * 2);  // [query | ctx]
  _Float16* src16 = (_Float16*)alloc(1024ull * 1024 * 2);
  _Float16* trg16 = (_Float16*)alloc(1024ull * 1024 * 2);
  _Float16* q16   = (_Float16*)alloc(2048ull * 512 * 2);
  _Float16* skey16= (_Float16*)alloc(1024ull * 512 * 2);
  _Float16* tkey16= (_Float16*)alloc(1024ull * 512 * 2);
  _Float16* sv16  = (_Float16*)alloc(1024ull * 512 * 2);
  _Float16* tv16  = (_Float16*)alloc(1024ull * 512 * 2);
  _Float16* tvT   = (_Float16*)alloc(16ull * 512 * 64 * 2);
  _Float16* WqT   = (_Float16*)alloc(512ull * 1024 * 2);
  _Float16* WsT   = (_Float16*)alloc(512ull * 1024 * 2);
  _Float16* WtT   = (_Float16*)alloc(512ull * 1024 * 2);
  _Float16* WsvT  = (_Float16*)alloc(512ull * 1024 * 2);
  _Float16* WtvT  = (_Float16*)alloc(512ull * 1024 * 2);
  _Float16* WoT   = (_Float16*)alloc(512ull * 1536 * 2);

  const int T = 256;
  // activations -> f16 (query goes into x16 cols [0,1024) with ld 1536)
  k_cvt_f16<<<(2048 * 1024 + T - 1) / T, T, 0, stream>>>(query, x16, 2048, 1024, 1536);
  k_cvt_f16<<<(1024 * 1024 + T - 1) / T, T, 0, stream>>>(src, src16, 1024, 1024, 1024);
  k_cvt_f16<<<(1024 * 1024 + T - 1) / T, T, 0, stream>>>(trg, trg16, 1024, 1024, 1024);
  // weights -> f16, transposed to [N][K]
  k_transpose_w<<<(1024 * 512 + T - 1) / T, T, 0, stream>>>(Wq,  WqT,  1024, 512);
  k_transpose_w<<<(1024 * 512 + T - 1) / T, T, 0, stream>>>(Ws,  WsT,  1024, 512);
  k_transpose_w<<<(1024 * 512 + T - 1) / T, T, 0, stream>>>(Wt,  WtT,  1024, 512);
  k_transpose_w<<<(1024 * 512 + T - 1) / T, T, 0, stream>>>(Wsv, WsvT, 1024, 512);
  k_transpose_w<<<(1024 * 512 + T - 1) / T, T, 0, stream>>>(Wtv, WtvT, 1024, 512);
  k_transpose_w<<<(1536 * 512 + T - 1) / T, T, 0, stream>>>(Wo,  WoT,  1536, 512);

  // projections (N=512 -> grid.x = 4; block tile 64x128)
  dim3 blk(128);
  k_gemm<<<dim3(4, 32), blk, 0, stream>>>(x16,   1536, WqT,  1024, bq,  q16,    nullptr, 512, 1024, 0);
  k_gemm<<<dim3(4, 16), blk, 0, stream>>>(src16, 1024, WsT,  1024, bs,  skey16, nullptr, 512, 1024, 0);
  k_gemm<<<dim3(4, 16), blk, 0, stream>>>(trg16, 1024, WtT,  1024, bt,  tkey16, nullptr, 512, 1024, 0);
  k_gemm<<<dim3(4, 16), blk, 0, stream>>>(src16, 1024, WsvT, 1024, bsv, sv16,   nullptr, 512, 1024, 0);
  k_gemm<<<dim3(4, 16), blk, 0, stream>>>(trg16, 1024, WtvT, 1024, btv, tv16,   nullptr, 512, 1024, 0);

  k_transpose_tv<<<(16 * 512 * 64 + T - 1) / T, T, 0, stream>>>(tv16, tvT);

  // fused trilinear attention (one block per (b,l))
  k_attn<<<dim3(128, 16), blk, 0, stream>>>(q16, skey16, tkey16, sv16, tvT, x16);

  // out = relu([query|ctx] @ Wo + bo), f32
  k_gemm<<<dim3(4, 32), blk, 0, stream>>>(x16, 1536, WoT, 1536, bo, nullptr, out, 512, 1536, 1);
}
